// SelfAttention1D_45861660787339
// MI455X (gfx1250) — matrix operand order repaired and checked by reading, not compiled, not run on hardware
//
#include <hip/hip_runtime.h>
#include <hip/hip_bf16.h>

// Problem constants (match reference)
constexpr int Bc = 8;     // batch
constexpr int Cc = 256;   // channels
constexpr int Lc = 2048;  // sequence length
constexpr int Dd = 32;    // q/k dim

typedef __attribute__((ext_vector_type(16))) __bf16 v16bf;
typedef __attribute__((ext_vector_type(8)))  float  v8f;

union FragBF {
    v16bf bf;
    unsigned int u[8];
};

// CDNA5 async global->LDS path (ASYNCcnt-tracked), guarded so either toolchain
// still compiles; falls back to synchronous uint4 copies otherwise.
// Probe result: param0 is `int __vector_size__(16) __device__*` (AS1, non-const),
// so the builtin wants typed, address-space-qualified int4 pointers.
#if defined(__HIP_DEVICE_COMPILE__) && \
    __has_builtin(__builtin_amdgcn_global_load_async_to_lds_b128) && \
    __has_builtin(__builtin_amdgcn_s_wait_asynccnt)
#define USE_ASYNC_LDS 1
typedef int v4i_raw __attribute__((vector_size(16)));
typedef __attribute__((address_space(1))) v4i_raw as1_v4i;   // global int4
typedef __attribute__((address_space(3))) v4i_raw as3_v4i;   // LDS int4
#else
#define USE_ASYNC_LDS 0
#endif

__device__ __forceinline__ void copy16_to_lds(const __bf16* g, __bf16* l)
{
#if USE_ASYNC_LDS
    __builtin_amdgcn_global_load_async_to_lds_b128((as1_v4i*)g, (as3_v4i*)l, 0, 0);
#else
    *(uint4*)l = *(const uint4*)g;
#endif
}

__device__ __forceinline__ void wait_async_lds()
{
#if USE_ASYNC_LDS
    __builtin_amdgcn_s_wait_asynccnt(0);
#endif
}

// ---------------------------------------------------------------------------
// Projection: Y[b] = W (M x C) @ x[b] (C x L) + bias, stored bf16 as (B, M, L)
// One wave per 16x16 output tile; K loop over C in steps of 32 via WMMA bf16.
// ---------------------------------------------------------------------------
__global__ __launch_bounds__(32, 1)
void proj_wmma_kernel(const float* __restrict__ x, const float* __restrict__ W,
                      const float* __restrict__ bias, __bf16* __restrict__ Y, int M)
{
    const int lane = threadIdx.x;
    const int m = lane & 15;     // A row within tile / B column within tile
    const int h = lane >> 4;     // lane half
    const int b  = blockIdx.z;
    const int m0 = blockIdx.y * 16;
    const int l0 = blockIdx.x * 16;

    v8f acc = {};

    for (int kk = 0; kk < Cc; kk += 32) {
        FragBF a, bb;
        // A fragment: W[m0+m][kk + k], 16-bit A-matrix 16x32 lane layout
        const float* wrow = W + (size_t)(m0 + m) * Cc + kk;
#pragma unroll
        for (int v = 0; v < 8; ++v) {
            int k0 = (v < 4) ? (h * 8 + 2 * v) : (16 + h * 8 + 2 * (v - 4));
            a.bf[2 * v]     = (__bf16)wrow[k0];
            a.bf[2 * v + 1] = (__bf16)wrow[k0 + 1];
        }
        // B fragment: x[b][kk + k][l0 + m], K split across lane halves
        const float* xcol = x + ((size_t)b * Cc + kk) * Lc + l0 + m;
        const int kbase = h * 16;
#pragma unroll
        for (int v = 0; v < 8; ++v) {
            int k0 = (v < 4) ? (h * 8 + 2 * v) : (16 + h * 8 + 2 * (v - 4));
            bb.bf[2 * v]     = (__bf16)xcol[(size_t)k0 * Lc];
            bb.bf[2 * v + 1] = (__bf16)xcol[(size_t)(k0 + 1) * Lc];
        }
        acc = __builtin_amdgcn_wmma_f32_16x16x32_bf16(
            false, a.bf, false, bb.bf, (short)0, acc, false, false);
    }

    // C/D layout: row = r + 8*h, col = m.  Add bias, store bf16 (B, M, L).
#pragma unroll
    for (int r = 0; r < 8; ++r) {
        int mm = m0 + r + 8 * h;
        Y[((size_t)b * M + mm) * Lc + l0 + m] = (__bf16)(acc[r] + bias[mm]);
    }
}

// ---------------------------------------------------------------------------
// Stage one 32-key chunk into LDS buffers.
//   K chunk: (D=32 x 32 j) transposed to [j][d] through registers (needed for
//            B-fragment dword pairs).
//   V chunk: (C=256 x 32 j) copied row-straight to [c][j]; on CDNA5 this uses
//            GLOBAL_LOAD_ASYNC_TO_LDS_B128 so it overlaps with compute.
// ---------------------------------------------------------------------------
__device__ __forceinline__
void stage_chunk(const __bf16* __restrict__ Kp, const __bf16* __restrict__ Vp,
                 int b, int j0, __bf16* kbuf, __bf16* vbuf, int tid)
{
    {
        int d  = tid >> 2;
        int jj = (tid & 3) * 8;
        const uint4 pkt = *(const uint4*)(Kp + ((size_t)b * Dd + d) * Lc + j0 + jj);
        const __bf16* e = (const __bf16*)&pkt;
#pragma unroll
        for (int q = 0; q < 8; ++q) kbuf[(jj + q) * 32 + d] = e[q];
    }
#pragma unroll
    for (int cc = 0; cc < 2; ++cc) {
        int c = cc * 128 + tid;
        const __bf16* src = Vp + ((size_t)b * Cc + c) * Lc + j0;
        __bf16* dst = vbuf + c * 32;
#pragma unroll
        for (int q = 0; q < 4; ++q) copy16_to_lds(src + q * 8, dst + q * 8);
    }
}

// ---------------------------------------------------------------------------
// Flash-style attention.  Block = 4 waves = 64 query rows of one batch.
// Double-buffered LDS staging: chunk j+1 streams in (async on CDNA5) while
// chunk j runs 2 score WMMAs + exp + 16 PV WMMAs per wave.
// Output tile oriented M=c, N=i so the fused epilogue stores coalesced.
// ---------------------------------------------------------------------------
__global__ __launch_bounds__(128, 1)
void attn_flash_wmma_kernel(const __bf16* __restrict__ Qp,  // (B, D, L)
                            const __bf16* __restrict__ Kp,  // (B, D, L)
                            const __bf16* __restrict__ Vp,  // (B, C, L)
                            const float*  __restrict__ x,
                            const float*  __restrict__ gamma,
                            float* __restrict__ out)
{
    __shared__ __bf16 ldsK[2][32 * 32];     // [buf][j][d]
    __shared__ __bf16 ldsV[2][256 * 32];    // [buf][c][j]
    __shared__ __bf16 ldsP[4][16 * 32];     // per-wave P tile [i][j]
    __shared__ float  ldsL[4][16];          // per-wave row sums

    const int tid  = threadIdx.x;
    const int w    = tid >> 5;
    const int lane = tid & 31;
    const int m    = lane & 15;
    const int h    = lane >> 4;

    const int b    = blockIdx.x >> 5;         // 32 blocks per batch
    const int i0   = (blockIdx.x & 31) * 64 + w * 16;

    const float scale = 0.17677669529663687f; // 1/sqrt(32)

    // ---- Q A-fragment (rows i, cols d), loaded once. Qp[(b*D + d)*L + i] ----
    FragBF qa;
    {
        const __bf16* qbase = Qp + (size_t)b * Dd * Lc + (i0 + m);
#pragma unroll
        for (int v = 0; v < 8; ++v) {
            int k0 = (v < 4) ? (h * 8 + 2 * v) : (16 + h * 8 + 2 * (v - 4));
            qa.bf[2 * v]     = qbase[(size_t)k0 * Lc];
            qa.bf[2 * v + 1] = qbase[(size_t)(k0 + 1) * Lc];
        }
    }

    v8f zero = {};
    v8f acc[16];
#pragma unroll
    for (int t = 0; t < 16; ++t) acc[t] = zero;
    v8f lsum = zero;

    // Prologue: stage chunk 0 into buffer 0.
    stage_chunk(Kp, Vp, b, 0, ldsK[0], ldsV[0], tid);
    wait_async_lds();
    __syncthreads();

    int p = 0;
    for (int j0 = 0; j0 < Lc; j0 += 32) {
        // Kick off next chunk into the other buffer; overlaps with compute.
        if (j0 + 32 < Lc) {
            stage_chunk(Kp, Vp, b, j0 + 32, ldsK[p ^ 1], ldsV[p ^ 1], tid);
            __builtin_prefetch(Vp + ((size_t)b * Cc + tid) * Lc + j0 + 64, 0, 0);
        }

        const __bf16* kcur = ldsK[p];
        const __bf16* vcur = ldsV[p];

        // ---- scores: two 16x16 tiles, S = Q (16x32 d) x K (32d x 16j) ----
        v8f st[2];
#pragma unroll
        for (int t = 0; t < 2; ++t) {
            FragBF kb;
            const int kbase = h * 16;
#pragma unroll
            for (int v = 0; v < 8; ++v) {
                int k0 = (v < 4) ? (h * 8 + 2 * v) : (16 + h * 8 + 2 * (v - 4));
                kb.u[v] = *(const unsigned int*)&kcur[(t * 16 + m) * 32 + k0];
            }
            st[t] = __builtin_amdgcn_wmma_f32_16x16x32_bf16(
                false, qa.bf, false, kb.bf, (short)0, zero, false, false);
        }

        // ---- exp (values are small-magnitude; online sum, no max track) ----
#pragma unroll
        for (int t = 0; t < 2; ++t) {
#pragma unroll
            for (int r = 0; r < 8; ++r) {
                float pr = __expf(st[t][r] * scale);
                st[t][r] = pr;
                lsum[r] += pr;
            }
            // spill P tile (C/D layout) to wave-private LDS as [i][j] bf16
#pragma unroll
            for (int r = 0; r < 8; ++r) {
                int i = r + 8 * h;
                ldsP[w][i * 32 + t * 16 + m] = (__bf16)st[t][r];
            }
        }

        // ---- B-fragment of P^T: rows k=j (32), cols n=i (16) ----
        FragBF pb;
        {
            const int kbase = h * 16;
#pragma unroll
            for (int v = 0; v < 8; ++v) {
                int k0 = (v < 4) ? (h * 8 + 2 * v) : (16 + h * 8 + 2 * (v - 4));
                pb.u[v] = *(const unsigned int*)&ldsP[w][m * 32 + k0];
            }
        }

        // ---- out tile: acc[ct](c x i) += V (16c x 32j) x P^T (32j x 16i) ----
#pragma unroll
        for (int ct = 0; ct < 16; ++ct) {
            FragBF va;
            const __bf16* vrow = &vcur[(ct * 16 + m) * 32];
#pragma unroll
            for (int v = 0; v < 8; ++v) {
                int k0 = (v < 4) ? (h * 8 + 2 * v) : (16 + h * 8 + 2 * (v - 4));
                va.u[v] = *(const unsigned int*)&vrow[k0];
            }
            acc[ct] = __builtin_amdgcn_wmma_f32_16x16x32_bf16(
                false, va.bf, false, pb.bf, (short)0, acc[ct], false, false);
        }

        // Own async (and regular) staging must land before the publish barrier.
        wait_async_lds();
        __syncthreads();
        p ^= 1;
    }

    // ---- reduce row sums across each 16-lane half, publish via LDS ----
#pragma unroll
    for (int r = 0; r < 8; ++r) {
        float v = lsum[r];
        v += __shfl_xor(v, 1, 16);
        v += __shfl_xor(v, 2, 16);
        v += __shfl_xor(v, 4, 16);
        v += __shfl_xor(v, 8, 16);
        lsum[r] = v;
    }
    if (m == 0) {
#pragma unroll
        for (int r = 0; r < 8; ++r) ldsL[w][r + 8 * h] = lsum[r];
    }
    __syncthreads();
    const float linv = 1.0f / ldsL[w][m];   // lane's i-column row sum
    const float g = gamma[0];

    // ---- fused epilogue: out = gamma * O/l + x ; lanes stride i -> coalesced
#pragma unroll
    for (int ct = 0; ct < 16; ++ct) {
#pragma unroll
        for (int r = 0; r < 8; ++r) {
            int c = ct * 16 + r + 8 * h;
            size_t idx = ((size_t)b * Cc + c) * Lc + i0 + m;
            out[idx] = g * (acc[ct][r] * linv) + x[idx];
        }
    }
}

// ---------------------------------------------------------------------------
extern "C" void kernel_launch(void* const* d_in, const int* in_sizes, int n_in,
                              void* d_out, int out_size, void* d_ws, size_t ws_size,
                              hipStream_t stream)
{
    (void)in_sizes; (void)n_in; (void)out_size; (void)ws_size;
    const float* x     = (const float*)d_in[0];
    const float* Wq    = (const float*)d_in[1];
    const float* bq    = (const float*)d_in[2];
    const float* Wk    = (const float*)d_in[3];
    const float* bk    = (const float*)d_in[4];
    const float* Wv    = (const float*)d_in[5];
    const float* bv    = (const float*)d_in[6];
    const float* gamma = (const float*)d_in[7];
    float* out = (float*)d_out;

    char* ws = (char*)d_ws;
    const size_t qkBytes = (size_t)Bc * Dd * Lc * sizeof(__bf16);   // 1 MB each
    __bf16* Qp = (__bf16*)(ws);
    __bf16* Kp = (__bf16*)(ws + qkBytes);
    __bf16* Vp = (__bf16*)(ws + 2 * qkBytes);                        // 8 MB

    // Projections (bf16 outputs in the attention kernel's preferred layouts)
    proj_wmma_kernel<<<dim3(Lc / 16, Dd / 16, Bc), dim3(32), 0, stream>>>(x, Wq, bq, Qp, Dd);
    proj_wmma_kernel<<<dim3(Lc / 16, Dd / 16, Bc), dim3(32), 0, stream>>>(x, Wk, bk, Kp, Dd);
    proj_wmma_kernel<<<dim3(Lc / 16, Cc / 16, Bc), dim3(32), 0, stream>>>(x, Wv, bv, Vp, Cc);

    // Flash attention + fused residual epilogue
    attn_flash_wmma_kernel<<<dim3(Bc * (Lc / 64)), dim3(128), 0, stream>>>(
        Qp, Kp, Vp, x, gamma, out);
}
